// Encoder_54692113547689
// MI455X (gfx1250) — matrix operand order repaired and checked by reading, not compiled, not run on hardware
//
#include <hip/hip_runtime.h>

// ---------------------------------------------------------------------------
// 3-layer GCN (DGL GraphConv, norm='both') for MI455X / gfx1250.
//   per layer: agg[dst] += x[src]*norm_src[src]  (edge scatter, L2-resident)
//              out = relu( (agg*norm_dst) @ W + b )   via v_wmma_f32_16x16x32_f16
//   final: column mean over nodes -> d_out[128]
// Workspace: bufA (N*D f32, agg), bufB (N*D f32, layer in/out), norms, packed W.
// ---------------------------------------------------------------------------

typedef __attribute__((ext_vector_type(16))) _Float16 v16h;
typedef __attribute__((ext_vector_type(8)))  float    v8f;

#define D 128

// ---------------- utility kernels ----------------

__global__ __launch_bounds__(256) void zero_f32(float* __restrict__ p, int n) {
  int i = blockIdx.x * 256 + threadIdx.x;
  if (i < n) p[i] = 0.0f;
}

__global__ __launch_bounds__(256) void degree_k(const int* __restrict__ src,
                                                const int* __restrict__ dst,
                                                float* __restrict__ deg_out,
                                                float* __restrict__ deg_in, int E) {
  int e = blockIdx.x * 256 + threadIdx.x;
  if (e < E) {
    atomicAdd(&deg_out[src[e]], 1.0f);
    atomicAdd(&deg_in[dst[e]], 1.0f);
  }
}

__global__ __launch_bounds__(256) void norm_k(float* __restrict__ a,
                                              float* __restrict__ b, int n) {
  int i = blockIdx.x * 256 + threadIdx.x;
  if (i < n) {
    a[i] = rsqrtf(fmaxf(a[i], 1.0f));
    b[i] = rsqrtf(fmaxf(b[i], 1.0f));
  }
}

// ---------------- SpMM: one wave per edge, 4 floats per lane ----------------
// Both x[src] gather and agg[dst] atomic scatter stay in the 192MB L2.

__global__ __launch_bounds__(256) void spmm_k(const int* __restrict__ src,
                                              const int* __restrict__ dst,
                                              const float* __restrict__ x,
                                              const float* __restrict__ norm_src,
                                              float* __restrict__ agg, int E) {
  int t = blockIdx.x * 256 + threadIdx.x;
  int e = t >> 5;
  if (e >= E) return;
  int lane = t & 31;
  int s = src[e];
  int d = dst[e];
  float ns = norm_src[s];
  float4 v = ((const float4*)(x + (size_t)s * D))[lane];
  float* a = agg + (size_t)d * D + lane * 4;
  atomicAdd(a + 0, v.x * ns);
  atomicAdd(a + 1, v.y * ns);
  atomicAdd(a + 2, v.z * ns);
  atomicAdd(a + 3, v.w * ns);
}

// ---------------- pack W (f32 row-major KxN) into B-fragment order ----------
// ISA 16-bit B layout (32x16 per 16x16x32 step): N = lane&15,
//   lanes 0-15 hold K=0..7,16..23, lanes 16-31 K=8..15,24..31; VGPR v: K=2(v&3)+16(v>>2),+1.
// Stored as [nchunk(8)][kb(4)][lane(32)][h(16)] halves -> contiguous 32B/lane.

__global__ __launch_bounds__(256) void pack_w_k(const float* __restrict__ W,
                                                _Float16* __restrict__ wp) {
  int idx = blockIdx.x * 256 + threadIdx.x;   // 0 .. 16383
  if (idx >= D * D) return;
  int h    = idx & 15;
  int lane = (idx >> 4) & 31;
  int kb   = (idx >> 9) & 3;
  int nc   = idx >> 11;
  int v    = h >> 1;
  int hilo = h & 1;
  int K = kb * 32 + ((v & 4) ? 16 : 0) + ((lane & 16) ? 8 : 0) + 2 * (v & 3) + hilo;
  int N = nc * 16 + (lane & 15);
  wp[idx] = (_Float16)W[K * D + N];
}

// ---------------- GEMM: one wave -> 16(M) x 128(N) strip, K=128 ------------
// A fragment built in-registers from agg row, fused *norm_dst and f32->f16.
// ISA 16-bit A 16x32 layout: M = lane&15; VGPR v in 0..3 -> K = 2v,2v+1
// (+8 for lanes 16-31); v in 4..7 adds +16.

__global__ __launch_bounds__(256) void gemm_wmma_k(const float* __restrict__ agg,
                                                   const float* __restrict__ norm_dst,
                                                   const _Float16* __restrict__ wp,
                                                   const float* __restrict__ bias,
                                                   float* __restrict__ out,
                                                   int do_relu, int n_tiles) {
  int wave = blockIdx.x * (blockDim.x >> 5) + (threadIdx.x >> 5);
  if (wave >= n_tiles) return;            // wave-uniform: EXEC all-1s for WMMA
  int lane = threadIdx.x & 31;
  int m0 = wave * 16;
  int mrow = m0 + (lane & 15);
  float nd = norm_dst[mrow];
  const float2* arow = (const float2*)(agg + (size_t)mrow * D);
  int khi = (lane & 16) ? 8 : 0;

  v8f acc[8] = {};

  #pragma unroll
  for (int kb = 0; kb < 4; ++kb) {
    v16h a;
    #pragma unroll
    for (int v = 0; v < 8; ++v) {
      int K = kb * 32 + ((v & 4) ? 16 : 0) + khi + 2 * (v & 3);
      float2 f2 = arow[K >> 1];
      a[2 * v]     = (_Float16)(f2.x * nd);
      a[2 * v + 1] = (_Float16)(f2.y * nd);
    }
    #pragma unroll
    for (int n = 0; n < 8; ++n) {
      v16h b = *(const v16h*)(wp + (((size_t)(n * 4 + kb) * 32 + lane) * 16));
      acc[n] = __builtin_amdgcn_wmma_f32_16x16x32_f16(
          /*neg_a=*/false, a, /*neg_b=*/false, b,
          /*c_mod=*/(short)0, acc[n], /*reuse_a=*/false, /*reuse_b=*/false);
    }
  }

  // C/D layout: VGPR r -> M = r (lanes 0-15) / 8+r (lanes 16-31); N = lane&15.
  int col0  = lane & 15;
  int rbase = (lane & 16) ? 8 : 0;
  #pragma unroll
  for (int n = 0; n < 8; ++n) {
    int col = n * 16 + col0;
    float bv = bias[col];
    #pragma unroll
    for (int r = 0; r < 8; ++r) {
      int row = m0 + rbase + r;
      float val = acc[n][r] + bv;
      if (do_relu) val = fmaxf(val, 0.0f);
      out[(size_t)row * D + col] = val;
    }
  }
}

// ---------------- column mean over all nodes -> d_out[128] ------------------

__global__ __launch_bounds__(128) void mean_k(const float* __restrict__ x,
                                              float* __restrict__ out, int n_nodes) {
  int col = threadIdx.x;
  float s = 0.0f;
  for (int row = blockIdx.x; row < n_nodes; row += gridDim.x)
    s += x[(size_t)row * D + col];
  atomicAdd(&out[col], s * (1.0f / (float)n_nodes));
}

// ---------------------------------------------------------------------------

extern "C" void kernel_launch(void* const* d_in, const int* in_sizes, int n_in,
                              void* d_out, int out_size, void* d_ws, size_t ws_size,
                              hipStream_t stream) {
  const float* h   = (const float*)d_in[0];
  const int*   src = (const int*)d_in[1];
  const int*   dst = (const int*)d_in[2];
  const float* W[3] = {(const float*)d_in[3], (const float*)d_in[5], (const float*)d_in[7]};
  const float* B[3] = {(const float*)d_in[4], (const float*)d_in[6], (const float*)d_in[8]};
  float* out = (float*)d_out;

  const int N = in_sizes[0] / D;     // 100000
  const int E = in_sizes[1];         // 1600000
  const size_t ND = (size_t)N * D;

  // workspace layout (d_ws is 256B aligned; all offsets keep 32B alignment)
  float*    bufA = (float*)d_ws;          // agg          N*D f32
  float*    bufB = bufA + ND;             // layer in/out N*D f32
  float*    nsrc = bufB + ND;             // norm_src     N f32
  float*    ndst = nsrc + N;              // norm_dst     N f32
  _Float16* wp   = (_Float16*)(ndst + N); // packed W     D*D f16

  const int n_tiles = N / 16;             // 6250, exact
  dim3 blk256(256), blk128(128);
  int gN   = (N + 255) / 256;
  int gND  = (int)((ND + 255) / 256);
  int gE   = (E + 255) / 256;
  int gE32 = (int)(((size_t)E * 32 + 255) / 256);
  int gGem = (n_tiles + 7) / 8;           // 8 waves per 256-thread block
  int gPk  = (D * D + 255) / 256;

  // degrees -> norms
  zero_f32<<<gN, blk256, 0, stream>>>(nsrc, N);
  zero_f32<<<gN, blk256, 0, stream>>>(ndst, N);
  degree_k<<<gE, blk256, 0, stream>>>(src, dst, nsrc, ndst, E);
  norm_k<<<gN, blk256, 0, stream>>>(nsrc, ndst, N);

  const float* x_in = h;
  for (int layer = 0; layer < 3; ++layer) {
    zero_f32<<<gND, blk256, 0, stream>>>(bufA, (int)ND);
    spmm_k<<<gE32, blk256, 0, stream>>>(src, dst, x_in, nsrc, bufA, E);
    pack_w_k<<<gPk, blk256, 0, stream>>>(W[layer], wp);
    gemm_wmma_k<<<gGem, blk256, 0, stream>>>(bufA, ndst, wp, B[layer], bufB,
                                             /*relu=*/(layer < 2) ? 1 : 0, n_tiles);
    x_in = bufB;
  }

  zero_f32<<<1, blk128, 0, stream>>>(out, D);
  mean_k<<<1024, blk128, 0, stream>>>(bufB, out, N);
}
